// Transformer_5695126634634
// MI455X (gfx1250) — matrix operand order repaired and checked by reading, not compiled, not run on hardware
//
#include <hip/hip_runtime.h>
#include <hip/hip_bf16.h>
#include <math.h>

typedef __attribute__((ext_vector_type(16))) _Float16 v16h;
typedef __attribute__((ext_vector_type(2)))  __fp16   v2fp16;
typedef __attribute__((ext_vector_type(8)))  float    v8f;

#define V_ 32000
#define D_ 1024
#define H_ 16
#define K_ 64
#define F_ 4096
#define L_ 4
#define S_ 2048
#define B_ 2

union FragA { v16h v; unsigned int u[8]; };

__device__ __forceinline__ unsigned int pack2(float a, float b) {
    union { v2fp16 h; unsigned int u; } p;
    p.h = __builtin_amdgcn_cvt_pkrtz(a, b);   // v_cvt_pk_rtz_f16_f32
    return p.u;
}

// ---------------------------------------------------------------------------
// WMMA GEMM: C[M,N] = A[M,Kd] @ B[Kd,N] + bias (+res) (relu optional)
// 256 threads = 8 waves; block tile 128x128; wave tile 32x64 (2x4 wmma tiles)
// Double-buffered LDS staging (f32 -> f16), float4 global loads.
// OUT_HALF: write C as f16 (used for K/V projections feeding attention).
// ---------------------------------------------------------------------------
template <bool RELU, bool HAS_RES, bool OUT_HALF>
__global__ __launch_bounds__(256)
void gemm_wmma(const float* __restrict__ A, const float* __restrict__ Bm,
               const float* __restrict__ bias, const float* res,
               float* C, int M, int N, int Kd)
{
    __shared__ _Float16 As[2][128 * 40];   // 128 x 32, rows padded to 40
    __shared__ _Float16 Bs[2][32 * 132];   // 32 x 128, rows padded to 132

    const int tid  = threadIdx.x;
    const int lane = tid & 31;
    const int w    = tid >> 5;
    const int wm   = w >> 1;     // 0..3
    const int wn   = w & 1;      // 0..1
    const int half = lane >> 4;
    const int r    = lane & 15;

    const int m0 = blockIdx.y * 128;
    const int n0 = blockIdx.x * 128;

    // staging coords (float4 granularity)
    const int ar  = (tid >> 3);              // A rows (4 chunks of 32)
    const int ac4 = (tid & 7) << 2;          // col 0,4,..28
    const int bk  = (tid >> 5);              // B rows (4 chunks of 8)
    const int bc4 = (tid & 31) << 2;         // col 0,4,..124

    v8f zero = {};
    v8f acc[2][4];
    #pragma unroll
    for (int i = 0; i < 2; i++)
        #pragma unroll
        for (int j = 0; j < 4; j++) acc[i][j] = zero;

    // A-fragment K offsets (ISA 16-bit A layout)
    int kv[8];
    #pragma unroll
    for (int v = 0; v < 8; v++) kv[v] = 16 * (v >> 2) + 8 * half + 2 * (v & 3);

    float4 ra[4], rb[4];

    auto load_tile = [&](int k0) {
        #pragma unroll
        for (int i = 0; i < 4; i++) {
            ra[i] = *(const float4*)(A + (size_t)(m0 + ar + i * 32) * Kd + k0 + ac4);
            rb[i] = *(const float4*)(Bm + (size_t)(k0 + bk + i * 8) * N + n0 + bc4);
        }
    };
    auto store_tile = [&](int buf) {
        uint2* A2 = (uint2*)As[buf];
        uint2* B2 = (uint2*)Bs[buf];
        #pragma unroll
        for (int i = 0; i < 4; i++) {
            uint2 pa; pa.x = pack2(ra[i].x, ra[i].y); pa.y = pack2(ra[i].z, ra[i].w);
            A2[(ar + i * 32) * 10 + (ac4 >> 2)] = pa;
            uint2 pb; pb.x = pack2(rb[i].x, rb[i].y); pb.y = pack2(rb[i].z, rb[i].w);
            B2[(bk + i * 8) * 33 + (bc4 >> 2)] = pb;
        }
    };

    load_tile(0);
    store_tile(0);

    int buf = 0;
    for (int k0 = 0; k0 < Kd; k0 += 32, buf ^= 1) {
        __syncthreads();
        const bool has_next = (k0 + 32 < Kd);
        if (has_next) load_tile(k0 + 32);       // overlap with WMMA below
        if (k0 + 64 < Kd) {                      // warm L2 for k+2
            __builtin_prefetch(A + (size_t)(m0 + ar) * Kd + k0 + 64 + ac4, 0, 0);
            __builtin_prefetch(Bm + (size_t)(k0 + 64 + bk) * N + n0 + bc4, 0, 0);
        }

        const unsigned int* As32 = (const unsigned int*)As[buf];
        const unsigned int* Bs32 = (const unsigned int*)Bs[buf];

        FragA af[2], bfr[4];
        #pragma unroll
        for (int tm = 0; tm < 2; tm++) {
            int rb_ = wm * 32 + tm * 16 + r;
            #pragma unroll
            for (int v = 0; v < 8; v++) af[tm].u[v] = As32[rb_ * 20 + (kv[v] >> 1)];
        }
        #pragma unroll
        for (int tn = 0; tn < 4; tn++) {
            int nb = wn * 64 + tn * 16;
            #pragma unroll
            for (int v = 0; v < 8; v++) bfr[tn].u[v] = (unsigned int)((const unsigned short*)Bs[buf])[kv[v] * 132 + nb + r] | ((unsigned int)((const unsigned short*)Bs[buf])[(kv[v] + 1) * 132 + nb + r] << 16);
        }

        #pragma unroll
        for (int tm = 0; tm < 2; tm++)
            #pragma unroll
            for (int tn = 0; tn < 4; tn++)
                acc[tm][tn] = __builtin_amdgcn_wmma_f32_16x16x32_f16(
                    false, af[tm].v, false, bfr[tn].v, (short)0, acc[tm][tn],
                    false, false);

        if (has_next) store_tile(buf ^ 1);
    }

    // Epilogue: bias (+relu) (+residual); one 64-bit mul per tile, ptr steps
    #pragma unroll
    for (int tm = 0; tm < 2; tm++) {
        #pragma unroll
        for (int tn = 0; tn < 4; tn++) {
            const int col = n0 + wn * 64 + tn * 16 + r;
            const float bvv = bias[col];
            size_t base = (size_t)(m0 + wm * 32 + tm * 16 + 8 * half) * (size_t)N
                          + (size_t)col;
            #pragma unroll
            for (int j = 0; j < 8; j++) {
                float vout = acc[tm][tn][j] + bvv;
                if (RELU)    vout = fmaxf(vout, 0.0f);
                if (HAS_RES) vout += res[base];
                if (OUT_HALF) ((_Float16*)C)[base] = (_Float16)vout;
                else          C[base] = vout;
                base += (size_t)N;
            }
        }
    }
}

// ---------------------------------------------------------------------------
// Flash attention: one wave per 16-query tile; key tiles of 32; head dim 64.
// Q in f32 (loaded once, scale folded), K/V pre-converted to f16.
// ---------------------------------------------------------------------------
__global__ __launch_bounds__(128)
void attn_wmma(const float* __restrict__ Q, const _Float16* __restrict__ Kb,
               const _Float16* __restrict__ Vb, float* __restrict__ O)
{
    __shared__ _Float16 Pl[4][16 * 32];

    const int tid  = threadIdx.x;
    const int lane = tid & 31;
    const int wv   = tid >> 5;
    const int gid  = blockIdx.x * 4 + wv;   // 0..4095
    const int qt   = gid >> 5;              // 0..127 (same qt within a block)
    const int bh   = gid & 31;
    const int h    = bh & 15;
    const int b    = bh >> 4;
    const int half = lane >> 4;
    const int r    = lane & 15;
    const int qbase = qt * 16;

    unsigned int* P32 = (unsigned int*)Pl[wv];

    int kvq[8];
    #pragma unroll
    for (int v = 0; v < 8; v++) kvq[v] = 16 * (v >> 2) + 8 * half + 2 * (v & 3);

    // Q fragments for two 32-wide k-steps of head dim 64; 1/sqrt(64) folded in
    FragA aq[2];
    {
        const float* qp = Q + ((size_t)(b * S_ + qbase + r)) * D_ + h * K_;
        #pragma unroll
        for (int ks = 0; ks < 2; ks++)
            #pragma unroll
            for (int v = 0; v < 8; v++) {
                int kk = ks * 32 + kvq[v];
                aq[ks].u[v] = pack2(qp[kk] * 0.125f, qp[kk + 1] * 0.125f);
            }
    }

    v8f zero = {};
    v8f o[4];
    #pragma unroll
    for (int i = 0; i < 4; i++) o[i] = zero;
    float mrow[8], lrow[8];
    #pragma unroll
    for (int j = 0; j < 8; j++) { mrow[j] = -1e30f; lrow[j] = 0.0f; }

    for (int kt = 0; kt <= qbase + 15; kt += 32) {
        // ---- scores: S(16x32) = Qtile @ K^T, two 16x16 n-tiles ----
        v8f sc[2];
        #pragma unroll
        for (int nt = 0; nt < 2; nt++) {
            const unsigned short* kp = (const unsigned short*)Kb
                + ((size_t)(b * S_ + kt + nt * 16)) * D_ + h * K_;
            #pragma unroll
            for (int ks = 0; ks < 2; ks++) {
                FragA bk;
                int dimk = ks * 32;             // B-frag: lane r = key column, elements = head dim (table order)
                #pragma unroll
                for (int v = 0; v < 8; v++) {
                    unsigned int lo = kp[(size_t)r * D_ + dimk + kvq[v]];
                    unsigned int hi = kp[(size_t)r * D_ + dimk + kvq[v] + 1];
                    bk.u[v] = lo | (hi << 16);
                }
                sc[nt] = __builtin_amdgcn_wmma_f32_16x16x32_f16(
                    false, aq[ks].v, false, bk.v, (short)0,
                    (ks ? sc[nt] : zero), false, false);
            }
        }

        // ---- causal mask ----
        #pragma unroll
        for (int nt = 0; nt < 2; nt++)
            #pragma unroll
            for (int j = 0; j < 8; j++) {
                int key  = kt + nt * 16 + r;
                int rowq = qbase + j + 8 * half;
                if (key > rowq) sc[nt][j] = -1e30f;
            }

        // ---- streaming softmax update (row stats via 16-lane shfl tree) ----
        #pragma unroll
        for (int j = 0; j < 8; j++) {
            float t = fmaxf(sc[0][j], sc[1][j]);
            #pragma unroll
            for (int dd = 1; dd < 16; dd <<= 1) t = fmaxf(t, __shfl_xor(t, dd, 32));
            float mn   = fmaxf(mrow[j], t);
            float corr = __expf(mrow[j] - mn);
            float p0 = __expf(sc[0][j] - mn);
            float p1 = __expf(sc[1][j] - mn);
            sc[0][j] = p0; sc[1][j] = p1;
            float s = p0 + p1;
            #pragma unroll
            for (int dd = 1; dd < 16; dd <<= 1) s += __shfl_xor(s, dd, 32);
            lrow[j] = lrow[j] * corr + s;
            mrow[j] = mn;
            #pragma unroll
            for (int sl = 0; sl < 4; sl++) o[sl][j] *= corr;
        }

        // ---- P (C-layout) -> LDS -> A-layout fragment ----
        #pragma unroll
        for (int nt = 0; nt < 2; nt++)
            #pragma unroll
            for (int j = 0; j < 8; j++)
                Pl[wv][(j + 8 * half) * 32 + nt * 16 + r] = (_Float16)sc[nt][j];
        __threadfence_block();

        FragA ap;
        #pragma unroll
        for (int v = 0; v < 8; v++) ap.u[v] = P32[r * 16 + (kvq[v] >> 1)];

        // ---- O += P(16x32) @ V(32x64), four 16-wide dim slices ----
        const _Float16* vp0 = Vb + ((size_t)(b * S_ + kt)) * D_ + h * K_;
        #pragma unroll
        for (int sl = 0; sl < 4; sl++) {
            const _Float16* vp = vp0 + sl * 16;
            FragA bv;
            #pragma unroll
            for (int v = 0; v < 8; v++)
                bv.u[v] = (unsigned int)((const unsigned short*)vp)[(size_t)kvq[v] * D_ + r] | ((unsigned int)((const unsigned short*)vp)[(size_t)(kvq[v] + 1) * D_ + r] << 16);  // f16 pair along key
            o[sl] = __builtin_amdgcn_wmma_f32_16x16x32_f16(
                false, ap.v, false, bv.v, (short)0, o[sl], false, false);
        }
    }

    // normalize and store [B,S,H*K]
    #pragma unroll
    for (int j = 0; j < 8; j++) {
        float inv = 1.0f / lrow[j];
        #pragma unroll
        for (int sl = 0; sl < 4; sl++) o[sl][j] *= inv;
    }
    #pragma unroll
    for (int sl = 0; sl < 4; sl++)
        #pragma unroll
        for (int j = 0; j < 8; j++) {
            int row = qbase + j + 8 * half;
            O[((size_t)(b * S_ + row)) * D_ + h * K_ + sl * 16 + r] = o[sl][j];
        }
}

// ---------------------------------------------------------------------------
__global__ void embed_kernel(const int* __restrict__ tok,
                             const float* __restrict__ emb,
                             const float* __restrict__ pos,
                             float* __restrict__ x)
{
    int idx = blockIdx.x * 256 + threadIdx.x;   // < B*S*D
    int bs = idx >> 10;                         // D = 1024
    int d  = idx & 1023;
    int s  = bs & (S_ - 1);
    int t  = tok[bs];
    x[idx] = emb[(size_t)t * D_ + d] + pos[(size_t)s * D_ + d];
}

// ---------------------------------------------------------------------------
extern "C" void kernel_launch(void* const* d_in, const int* in_sizes, int n_in,
                              void* d_out, int out_size, void* d_ws, size_t ws_size,
                              hipStream_t stream)
{
    const int*   tokens = (const int*)d_in[0];
    const float* emb    = (const float*)d_in[1];
    const float* pos    = (const float*)d_in[2];
    const float* Wq = (const float*)d_in[3];
    const float* bq = (const float*)d_in[4];
    const float* Wk = (const float*)d_in[5];
    const float* bk = (const float*)d_in[6];
    const float* Wv = (const float*)d_in[7];
    const float* bv = (const float*)d_in[8];
    const float* Wo = (const float*)d_in[9];
    const float* bo = (const float*)d_in[10];
    const float* W1 = (const float*)d_in[11];
    const float* b1 = (const float*)d_in[12];
    const float* W2 = (const float*)d_in[13];
    const float* b2 = (const float*)d_in[14];
    const float* Wf = (const float*)d_in[15];
    const float* bfv = (const float*)d_in[16];
    float* out = (float*)d_out;

    const int M = B_ * S_;                 // 4096
    const size_t xN = (size_t)M * D_;      // 4M floats
    float* x  = (float*)d_ws;              // [0,   4M) f32
    float* q  = x  + xN;                   // [4M,  8M) f32
    _Float16* k16 = (_Float16*)(q + xN);   // 4M halves = 2M floats
    _Float16* v16 = (_Float16*)(q + xN + xN / 2);
    float* ao = q + 2 * xN;                // [12M, 16M) f32
    float* hb = ao + xN;                   // [16M, 32M) f32 (M*F)

    embed_kernel<<<(M * D_) / 256, 256, 0, stream>>>(tokens, emb, pos, x);

    dim3 gD(D_ / 128, M / 128);
    dim3 gF(F_ / 128, M / 128);
    dim3 gV(V_ / 128, M / 128);

    for (int l = 0; l < L_; l++) {
        const float* Wqi = Wq + (size_t)l * D_ * D_;
        const float* Wki = Wk + (size_t)l * D_ * D_;
        const float* Wvi = Wv + (size_t)l * D_ * D_;
        const float* Woi = Wo + (size_t)l * D_ * D_;
        const float* W1i = W1 + (size_t)l * D_ * F_;
        const float* W2i = W2 + (size_t)l * F_ * D_;
        const float* bqi = bq + (size_t)l * D_;
        const float* bki = bk + (size_t)l * D_;
        const float* bvi = bv + (size_t)l * D_;
        const float* boi = bo + (size_t)l * D_;
        const float* b1i = b1 + (size_t)l * F_;
        const float* b2i = b2 + (size_t)l * D_;

        gemm_wmma<false, false, false><<<gD, 256, 0, stream>>>(
            x, Wqi, bqi, nullptr, q, M, D_, D_);
        gemm_wmma<false, false, true ><<<gD, 256, 0, stream>>>(
            x, Wki, bki, nullptr, (float*)k16, M, D_, D_);
        gemm_wmma<false, false, true ><<<gD, 256, 0, stream>>>(
            x, Wvi, bvi, nullptr, (float*)v16, M, D_, D_);

        attn_wmma<<<(B_ * H_ * (S_ / 16)) / 4, 128, 0, stream>>>(q, k16, v16, ao);

        gemm_wmma<false, true,  false><<<gD, 256, 0, stream>>>(
            ao, Woi, boi, x, x, M, D_, D_);
        gemm_wmma<true,  false, false><<<gF, 256, 0, stream>>>(
            x, W1i, b1i, nullptr, hb, M, F_, D_);
        gemm_wmma<false, true,  false><<<gD, 256, 0, stream>>>(
            hb, W2i, b2i, x, x, M, D_, F_);
    }

    gemm_wmma<false, false, false><<<gV, 256, 0, stream>>>(
        x, Wf, bfv, nullptr, out, M, V_, D_);
}
